// _NVFP4Block_42649025249845
// MI455X (gfx1250) — compile-verified, matrix operand order checked by reading, not run on hardware
//
#include <hip/hip_runtime.h>
#include <hip/hip_bf16.h>

// ---------------------------------------------------------------------------
// MLP block: LayerNorm -> fc1(+b1) -> GELU(exact) -> fc2(+b2)
// T=8192, H=4096, I=16384, fp32 in/out.  2.2 TFLOP vs <1GB traffic ->
// compute-bound: bf16 WMMA (v_wmma_f32_16x16x32_bf16) with f32 accumulate.
// Data movement: Tensor Data Mover (tensor_load_to_lds) with double-buffered
// LDS and TDM-generated padding (40-halfword row stride, bank-conflict free).
// Descriptors are kept fully scalar (readfirstlane'd wave id) and hoisted.
// ---------------------------------------------------------------------------

#define HIDDEN 4096
#define INTER  16384
#define LN_EPS 1e-5f

typedef unsigned short u16;
typedef unsigned int   u32;
typedef unsigned long long u64;

typedef __attribute__((ext_vector_type(16))) __bf16 v16bf;
typedef __attribute__((ext_vector_type(8)))  float  v8f;
typedef __attribute__((ext_vector_type(4)))  float  v4f;
typedef __attribute__((ext_vector_type(4)))  u32    v4u;
typedef __attribute__((ext_vector_type(8)))  u32    v8u;

union Frag {
    v16bf v;
    uint4 u[2];   // 32 bytes = 16 bf16
};

// float -> bf16, round-to-nearest-even
__device__ __forceinline__ u16 f2bf(float f) {
    u32 u = __float_as_uint(f);
    u32 r = u + 0x7FFFu + ((u >> 16) & 1u);
    return (u16)(r >> 16);
}

__device__ __forceinline__ uint4 pack8(const float* f) {
    uint4 o;
    o.x = (u32)f2bf(f[0]) | ((u32)f2bf(f[1]) << 16);
    o.y = (u32)f2bf(f[2]) | ((u32)f2bf(f[3]) << 16);
    o.z = (u32)f2bf(f[4]) | ((u32)f2bf(f[5]) << 16);
    o.w = (u32)f2bf(f[6]) | ((u32)f2bf(f[7]) << 16);
    return o;
}

__device__ __forceinline__ float gelu_exact(float x) {
    return 0.5f * x * (1.0f + erff(x * 0.70710678118654752f));
}

// ---------------------------------------------------------------------------
// TDM: DMA a [tileRows x 32] bf16 tile (row stride K elements) into LDS,
// inserting 16B of padding after every 64B row -> LDS stride 40 halfwords.
// D# per CDNA5 ISA 8.3/8.4.  Groups 2/3 unused (<=2D tensor), zeroed.
// ---------------------------------------------------------------------------
__device__ __forceinline__ v8u tdm_make_g1(int K, int tileRows) {
    v8u g1;
    g1[0] = (1u << 16)        // data_size = 2 bytes
          | (1u << 20)        // pad_enable
          | (3u << 22)        // pad_interval: 16 DWORDs (one 64B tile row)
          | (3u << 25);       // pad_amount: 4 DWORDs (16B)  -> stride 80B
    g1[1] = 32u << 16;                          // tensor_dim0 = 32 (lo16)
    g1[2] = ((u32)tileRows & 0xFFFFu) << 16;    // tensor_dim1 lo16
    g1[3] = 32u << 16;                          // tile_dim0 = 32
    g1[4] = (u32)tileRows;                      // tile_dim1 (tile_dim2=0)
    g1[5] = (u32)K;                             // tensor_dim0_stride lo32
    g1[6] = 0u;
    g1[7] = 0u;
    return g1;
}

__device__ __forceinline__ void tdm_issue(u32 ldsOff, u64 ga, v8u g1) {
    v4u g0;
    g0[0] = 1u;                                              // count=1, user D#
    g0[1] = ldsOff;                                          // lds_addr (bytes)
    g0[2] = (u32)ga;                                         // global_addr[31:0]
    g0[3] = (u32)((ga >> 32) & 0x01FFFFFFu) | (2u << 30);    // ga[56:32] | type=2
    v4u z = (v4u)(0u);
    asm volatile("tensor_load_to_lds %0, %1, %2, %3"
                 :: "s"(g0), "s"(g1), "s"(z), "s"(z)
                 : "memory");
}

// ---------------------------------------------------------------------------
// fp32 -> bf16 bulk convert (8 elements / thread), n % 8 == 0
// ---------------------------------------------------------------------------
__global__ __launch_bounds__(256) void cvt_bf16_kernel(const float* __restrict__ in,
                                                       u16* __restrict__ out, size_t n) {
    size_t i = ((size_t)blockIdx.x * blockDim.x + threadIdx.x) * 8;
    if (i >= n) return;
    v4f a = __builtin_nontemporal_load((const v4f*)(in + i));
    v4f b = __builtin_nontemporal_load((const v4f*)(in + i + 4));
    float f[8] = {a[0], a[1], a[2], a[3], b[0], b[1], b[2], b[3]};
    *(uint4*)(out + i) = pack8(f);
}

// ---------------------------------------------------------------------------
// LayerNorm over HIDDEN=4096, one row per 256-thread block, bf16 output
// ---------------------------------------------------------------------------
__global__ __launch_bounds__(256) void layernorm_kernel(const float* __restrict__ x,
                                                        const float* __restrict__ w,
                                                        const float* __restrict__ b,
                                                        u16* __restrict__ out) {
    const int row = blockIdx.x;
    const int tid = threadIdx.x;
    const float* xr = x + (size_t)row * HIDDEN;
    const int base = tid * 16;

    v4f v[4];
#pragma unroll
    for (int j = 0; j < 4; ++j)
        v[j] = __builtin_nontemporal_load((const v4f*)(xr + base + j * 4));

    float s = 0.f, sq = 0.f;
#pragma unroll
    for (int j = 0; j < 4; ++j)
#pragma unroll
        for (int e = 0; e < 4; ++e) {
            s  += v[j][e];
            sq += v[j][e] * v[j][e];
        }

    __shared__ float ssum[256], ssq[256];
    ssum[tid] = s; ssq[tid] = sq;
    __syncthreads();
#pragma unroll
    for (int off = 128; off > 0; off >>= 1) {
        if (tid < off) { ssum[tid] += ssum[tid + off]; ssq[tid] += ssq[tid + off]; }
        __syncthreads();
    }
    const float mean = ssum[0] * (1.0f / HIDDEN);
    const float var  = ssq[0]  * (1.0f / HIDDEN) - mean * mean;
    const float rstd = rsqrtf(var + LN_EPS);

    float y[16];
#pragma unroll
    for (int j = 0; j < 4; ++j) {
        v4f gw = *(const v4f*)(w + base + j * 4);
        v4f gb = *(const v4f*)(b + base + j * 4);
#pragma unroll
        for (int e = 0; e < 4; ++e)
            y[j * 4 + e] = (v[j][e] - mean) * rstd * gw[e] + gb[e];
    }
    u16* orow = out + (size_t)row * HIDDEN + base;
    *(uint4*)(orow)     = pack8(y);
    *(uint4*)(orow + 8) = pack8(y + 8);
}

// ---------------------------------------------------------------------------
// WMMA GEMM:  C[M,N] = A[M,K](bf16) * B[N,K](bf16)^T + bias[N]
// Block tile 256x128, BK=32, 256 threads = 8 waves (4 along M x 2 along N),
// each wave owns a 64x64 tile = 4x4 accumulators (16 WMMA per K-step).
// LDS fill: ONE TDM per wave per K-step — waves 0..3 DMA 64-row A slices,
// waves 4..7 DMA 32-row B slices.  Double-buffered; one barrier per K-step.
// ---------------------------------------------------------------------------
template <bool DO_GELU, bool OUT_BF16>
__global__ __launch_bounds__(256) void gemm_wmma_kernel(const u16* __restrict__ A,
                                                        const u16* __restrict__ B,
                                                        const float* __restrict__ bias,
                                                        void* __restrict__ Cout,
                                                        int M, int N, int K) {
    constexpr int BM = 256, BN = 128, BK = 32;
    constexpr int LDA = BK + 8;   // 40 halfwords = 80B: matches TDM padding

    __shared__ u16 As[2][BM][LDA];
    __shared__ u16 Bs[2][BN][LDA];

    const int tid  = threadIdx.x;
    const int lane = tid & 31;
    const int wave = tid >> 5;        // per-lane copy, used for VGPR addressing
    const int waveM = wave & 3;
    const int waveN = wave >> 2;
    const int half  = lane >> 4;
    const int l16   = lane & 15;

    const int rowBlk = blockIdx.y * BM;
    const int colBlk = blockIdx.x * BN;

    // ---- scalar (SGPR) TDM setup: one slice per wave --------------------
    const int swave = __builtin_amdgcn_readfirstlane(tid >> 5);
    const bool isA  = swave < 4;
    const int rows  = isA ? 64 : 32;

    u64 ga;
    u32 ldsOff0, ldsOff1;
    if (isA) {
        ga      = (u64)(size_t)(A + (size_t)(rowBlk + swave * 64) * K);
        ldsOff0 = (u32)(size_t)&As[0][swave * 64][0];
        ldsOff1 = (u32)(size_t)&As[1][swave * 64][0];
    } else {
        ga      = (u64)(size_t)(B + (size_t)(colBlk + (swave - 4) * 32) * K);
        ldsOff0 = (u32)(size_t)&Bs[0][(swave - 4) * 32][0];
        ldsOff1 = (u32)(size_t)&Bs[1][(swave - 4) * 32][0];
    }
    const v8u g1 = tdm_make_g1(K, rows);

    v8f acc[4][4] = {};

    int cur = 0;
    tdm_issue(ldsOff0, ga, g1);   // stage 0 into buffer 0

    for (int k0 = 0; k0 < K; k0 += BK) {
        __builtin_amdgcn_s_wait_tensorcnt(0);
        __syncthreads();   // all slices for buffer `cur` are in LDS

        if (k0 + BK < K)   // prefetch next stage into the other buffer
            tdm_issue(cur ? ldsOff0 : ldsOff1, ga + 64, g1);
        ga += 64;          // BK elements * 2 bytes

        // Fragments per ISA 16-bit A/B layout: lane holds row/col l16,
        // K chunks [half*8, +8) and [16+half*8, +8) -> two 16B LDS reads.
        Frag af[4], bfr[4];
        const int klo = half * 8;
#pragma unroll
        for (int mt = 0; mt < 4; ++mt) {
            const int r = waveM * 64 + mt * 16 + l16;
            af[mt].u[0] = *(const uint4*)&As[cur][r][klo];
            af[mt].u[1] = *(const uint4*)&As[cur][r][16 + klo];
        }
#pragma unroll
        for (int nt = 0; nt < 4; ++nt) {
            const int c = waveN * 64 + nt * 16 + l16;
            bfr[nt].u[0] = *(const uint4*)&Bs[cur][c][klo];
            bfr[nt].u[1] = *(const uint4*)&Bs[cur][c][16 + klo];
        }

#pragma unroll
        for (int mt = 0; mt < 4; ++mt)
#pragma unroll
            for (int nt = 0; nt < 4; ++nt)
                acc[mt][nt] = __builtin_amdgcn_wmma_f32_16x16x32_bf16(
                    false, af[mt].v, false, bfr[nt].v,
                    (short)0, acc[mt][nt], false, false);

        cur ^= 1;
    }

    // Epilogue. C/D layout: lane -> n = l16, m = vgpr + 8*half.
    float* outF = (float*)Cout;
    u16*   outB = (u16*)Cout;
#pragma unroll
    for (int mt = 0; mt < 4; ++mt) {
#pragma unroll
        for (int nt = 0; nt < 4; ++nt) {
            const int rBase = rowBlk + waveM * 64 + mt * 16 + half * 8;
            const int col   = colBlk + waveN * 64 + nt * 16 + l16;
            const float bv  = bias[col];
#pragma unroll
            for (int v = 0; v < 8; ++v) {
                float x = acc[mt][nt][v] + bv;
                if (DO_GELU) x = gelu_exact(x);
                const size_t idx = (size_t)(rBase + v) * N + col;
                if (OUT_BF16) __builtin_nontemporal_store(f2bf(x), outB + idx);
                else          outF[idx] = x;
            }
        }
    }
}

// ---------------------------------------------------------------------------
// Launch.  inputs: x, ln_w, ln_b, w1, b1, w2, b2
// ws: X_bf16 (64MB) | W1_bf16 (128MB) | W2_bf16 (128MB) | G_bf16 (256MB)
// ---------------------------------------------------------------------------
extern "C" void kernel_launch(void* const* d_in, const int* in_sizes, int n_in,
                              void* d_out, int out_size, void* d_ws, size_t ws_size,
                              hipStream_t stream) {
    const float* x    = (const float*)d_in[0];
    const float* ln_w = (const float*)d_in[1];
    const float* ln_b = (const float*)d_in[2];
    const float* w1   = (const float*)d_in[3];
    const float* b1   = (const float*)d_in[4];
    const float* w2   = (const float*)d_in[5];
    const float* b2   = (const float*)d_in[6];

    const int T = in_sizes[0] / HIDDEN;   // 8192

    u16* Xb  = (u16*)d_ws;
    u16* W1b = Xb  + (size_t)T * HIDDEN;
    u16* W2b = W1b + (size_t)INTER * HIDDEN;
    u16* Gb  = W2b + (size_t)HIDDEN * INTER;

    const size_t nW = (size_t)INTER * HIDDEN;

    {
        int blocks = (int)(nW / (8 * 256));
        cvt_bf16_kernel<<<blocks, 256, 0, stream>>>(w1, W1b, nW);
        cvt_bf16_kernel<<<blocks, 256, 0, stream>>>(w2, W2b, nW);
    }

    layernorm_kernel<<<T, 256, 0, stream>>>(x, ln_w, ln_b, Xb);

    {   // fc1 + bias + GELU -> bf16 [T, INTER]
        dim3 grid(INTER / 128, T / 256);
        gemm_wmma_kernel<true, true><<<grid, 256, 0, stream>>>(
            Xb, W1b, b1, (void*)Gb, T, INTER, HIDDEN);
    }
    {   // fc2 + bias -> fp32 output [T, HIDDEN]
        dim3 grid(HIDDEN / 128, T / 256);
        gemm_wmma_kernel<false, false><<<grid, 256, 0, stream>>>(
            Gb, W2b, b2, d_out, T, HIDDEN, INTER);
    }
    (void)n_in; (void)out_size; (void)ws_size;
}
